// TFSwinSelfAttention_77189152244335
// MI455X (gfx1250) — matrix operand order repaired and checked by reading, not compiled, not run on hardware
//
#include <hip/hip_runtime.h>

typedef _Float16 half_t;
typedef _Float16 v16h __attribute__((ext_vector_type(16)));
typedef _Float16 v8h  __attribute__((ext_vector_type(8)));
typedef float    v8f  __attribute__((ext_vector_type(8)));

#define WIN_N   49
#define NPAD    64
#define C_DIM   256
#define NHEADS  8
#define HDIM    32
#define SCALE   0.17677669529663687f   // 1/sqrt(32)

static __device__ __forceinline__ v16h cat16(v8h lo, v8h hi) {
    return __builtin_shufflevector(lo, hi, 0,1,2,3,4,5,6,7,8,9,10,11,12,13,14,15);
}

// ---------------------------------------------------------------------------
// Prep: (a) transpose+convert weights to f16 wt[mat][n][k]
//       (b) fuse rel-pos bias + attention mask into combined[w][h][64][64] f32
//           (pad cols k>=49 -> -1e30 so padded keys vanish in softmax)
// ---------------------------------------------------------------------------
__global__ void prep_kernel(const float* __restrict__ wq, const float* __restrict__ wk,
                            const float* __restrict__ wv,
                            const float* __restrict__ bias_table,
                            const int*   __restrict__ rpi,
                            const float* __restrict__ mask,
                            half_t* __restrict__ wt, float* __restrict__ combined) {
    const int WT_ELEMS = 3 * 256 * 256;
    int idx = blockIdx.x * 256 + threadIdx.x;
    if (idx < WT_ELEMS) {
        int mat = idx >> 16;
        int rem = idx & 65535;
        int n = rem >> 8;     // output column
        int k = rem & 255;    // input dim
        const float* w = (mat == 0) ? wq : (mat == 1) ? wk : wv;
        wt[idx] = (half_t)w[k * 256 + n];       // wt[mat][n][k] = W[k][n]
    } else {
        int j = idx - WT_ELEMS;
        if (j < 16 * 8 * 64 * 64) {
            int k = j & 63;
            int m = (j >> 6) & 63;
            int h = (j >> 12) & 7;
            int w = j >> 15;
            float v;
            if (k >= WIN_N)      v = -1e30f;
            else if (m >= WIN_N) v = 0.0f;
            else v = bias_table[rpi[m * WIN_N + k] * NHEADS + h]
                   + mask[(w * WIN_N + m) * WIN_N + k];
            combined[j] = v;
        }
    }
}

// ---------------------------------------------------------------------------
// Fused Swin window attention. 1 block = 1 window (grid 4096), 256 threads
// = 8 waves; wave w handles head w in the attention phase.
// LDS layout (halves): xs[64][256] | qs[8][64][32] | ks[8][64][32]
//                      | vs[8][32][64] (V transposed) | ps[8][16][64]
// ---------------------------------------------------------------------------
__global__ __launch_bounds__(256, 1)
void swin_attn_kernel(const float* __restrict__ X,
                      const half_t* __restrict__ wt_all,
                      const float* __restrict__ bq,
                      const float* __restrict__ bk,
                      const float* __restrict__ bv,
                      const float* __restrict__ combined,
                      float* __restrict__ out) {
    extern __shared__ char smem_raw[];
    half_t* xs = (half_t*)smem_raw;          // 16384 halves
    half_t* qs = xs + 64 * 256;              // 16384
    half_t* ks = qs + 8 * 64 * 32;           // 16384
    half_t* vs = ks + 8 * 64 * 32;           // 16384 ([h][d][key], transposed)
    half_t* ps = vs + 8 * 32 * 64;           // 8192

    const int b    = blockIdx.x;
    const int tid  = threadIdx.x;
    const int lane = tid & 31;
    const int wave = tid >> 5;
    const int lh   = lane & 15;   // position within 16-lane half
    const int hp   = lane >> 4;   // which half of the wave

    // ---- phase 1: stage window input as f16 into LDS (vectorized, pad=0) ----
    {
        const float* xb = X + (size_t)b * WIN_N * C_DIM;
        #pragma unroll
        for (int it = 0; it < 8; ++it) {                 // 2048 v8h chunks total
            int i    = tid + it * 256;
            int base = i * 8;
            int r    = base >> 8;
            v8h hv;
            if (r < WIN_N) {
                const float4 f0 = *(const float4*)(xb + base);
                const float4 f1 = *(const float4*)(xb + base + 4);
                hv[0] = (half_t)f0.x; hv[1] = (half_t)f0.y;
                hv[2] = (half_t)f0.z; hv[3] = (half_t)f0.w;
                hv[4] = (half_t)f1.x; hv[5] = (half_t)f1.y;
                hv[6] = (half_t)f1.z; hv[7] = (half_t)f1.w;
            } else {
                hv = v8h{};
            }
            *(v8h*)(xs + base) = hv;
        }
    }
    __syncthreads();

    // ---- phase 2: QKV GEMMs. Per wave: 3 matrices x 2 n-tiles (wave, wave+8),
    //      sharing each X A-fragment across both n-tiles. ----
    #pragma unroll
    for (int mat = 0; mat < 3; ++mat) {
        const half_t* wt   = wt_all + mat * 65536;
        const float*  bias = (mat == 0) ? bq : (mat == 1) ? bk : bv;
        const int nt0 = wave, nt1 = wave + 8;

        v8f acc0[4] = {v8f{}, v8f{}, v8f{}, v8f{}};
        v8f acc1[4] = {v8f{}, v8f{}, v8f{}, v8f{}};
        const half_t* wrow0 = wt + (nt0 * 16 + lh) * 256 + hp * 16;
        const half_t* wrow1 = wt + (nt1 * 16 + lh) * 256 + hp * 16;
        #pragma unroll
        for (int k8 = 0; k8 < 8; ++k8) {
            const int kb = k8 * 32;
            v16h b0 = *(const v16h*)(wrow0 + kb);
            v16h b1 = *(const v16h*)(wrow1 + kb);
            const half_t* xrow = xs + lh * 256 + kb + hp * 16;
            #pragma unroll
            for (int mt = 0; mt < 4; ++mt) {
                const half_t* p = xrow + mt * 16 * 256;
                v16h a = cat16(*(const v8h*)p, *(const v8h*)(p + 8));
                acc0[mt] = __builtin_amdgcn_wmma_f32_16x16x32_f16(
                    false, a, false, b0, (short)0, acc0[mt], false, false);
                acc1[mt] = __builtin_amdgcn_wmma_f32_16x16x32_f16(
                    false, a, false, b1, (short)0, acc1[mt], false, false);
            }
        }
        #pragma unroll
        for (int half = 0; half < 2; ++half) {
            const int   ncol = (half ? nt1 : nt0) * 16 + lh; // N = lane&15
            const float bval = bias[ncol];
            const int   h    = ncol >> 5;
            const int   d    = ncol & 31;
            v8f* acc = half ? acc1 : acc0;
            #pragma unroll
            for (int mt = 0; mt < 4; ++mt) {
                #pragma unroll
                for (int r = 0; r < 8; ++r) {
                    int   m = mt * 16 + r + hp * 8;
                    float v = acc[mt][r] + bval;
                    if (mat == 0)      qs[h * 2048 + m * 32 + d] = (half_t)v;
                    else if (mat == 1) ks[h * 2048 + m * 32 + d] = (half_t)v;
                    else               vs[h * 2048 + d * 64 + m] = (half_t)v;
                }
            }
        }
    }
    __syncthreads();

    // ---- phase 3: attention, one head per wave ----
    const int h = wave;
    const half_t* qsh = qs + h * 2048;
    const half_t* ksh = ks + h * 2048;
    const half_t* vsh = vs + h * 2048;
    half_t*       psw = ps + wave * 1024;
    const float*  cmb = combined + (size_t)((b & 15) * 8 + h) * 4096;
    float*        outb = out + (size_t)b * WIN_N * C_DIM + h * HDIM;

    // K^T B-fragments: reused by all 4 row tiles -> hoist
    v16h kf[4];
    #pragma unroll
    for (int ct = 0; ct < 4; ++ct)
        kf[ct] = *(const v16h*)(ksh + (ct * 16 + lh) * 32 + hp * 16);

    for (int rt = 0; rt < 4; ++rt) {
        // Q A-fragment (16x32, K = head dim)
        const half_t* qrow = qsh + (rt * 16 + lh) * 32 + hp * 16;
        v16h qa = cat16(*(const v8h*)qrow, *(const v8h*)(qrow + 8));

        v8f s[4];
        #pragma unroll
        for (int ct = 0; ct < 4; ++ct) {
            v8f z = {};
            s[ct] = __builtin_amdgcn_wmma_f32_16x16x32_f16(
                false, qa, false, kf[ct], (short)0, z, false, false);
        }
        // scale + fused rel-bias/mask (L2-resident, coalesced per half-wave)
        #pragma unroll
        for (int ct = 0; ct < 4; ++ct) {
            #pragma unroll
            for (int r = 0; r < 8; ++r) {
                int m = rt * 16 + r + hp * 8;
                s[ct][r] = s[ct][r] * SCALE + cmb[m * 64 + ct * 16 + lh];
            }
        }
        // row softmax: rows live across 16-lane halves (N = lane&15)
        #pragma unroll
        for (int r = 0; r < 8; ++r) {
            float mx = fmaxf(fmaxf(s[0][r], s[1][r]), fmaxf(s[2][r], s[3][r]));
            #pragma unroll
            for (int off = 1; off < 16; off <<= 1)
                mx = fmaxf(mx, __shfl_xor(mx, off, 32));
            float e0 = __expf(s[0][r] - mx), e1 = __expf(s[1][r] - mx);
            float e2 = __expf(s[2][r] - mx), e3 = __expf(s[3][r] - mx);
            float sm = e0 + e1 + e2 + e3;
            #pragma unroll
            for (int off = 1; off < 16; off <<= 1)
                sm += __shfl_xor(sm, off, 32);
            float rinv = __builtin_amdgcn_rcpf(sm);
            s[0][r] = e0 * rinv; s[1][r] = e1 * rinv;
            s[2][r] = e2 * rinv; s[3][r] = e3 * rinv;
        }
        // probs -> LDS f16 (row-tile only, 16x64)
        #pragma unroll
        for (int ct = 0; ct < 4; ++ct)
            #pragma unroll
            for (int r = 0; r < 8; ++r)
                psw[(r + hp * 8) * 64 + ct * 16 + lh] = (half_t)s[ct][r];
        __syncthreads();   // uniform trip count: makes cross-lane LDS visible

        // P A-fragments for the two 32-wide K chunks
        v16h pa[2];
        #pragma unroll
        for (int kc = 0; kc < 2; ++kc) {
            const half_t* prow = psw + lh * 64 + kc * 32 + hp * 16;
            pa[kc] = cat16(*(const v8h*)prow, *(const v8h*)(prow + 8));
        }
        // ctx = P.V : B[k][d] = vt[d][k] -> contiguous row read of vs
        #pragma unroll
        for (int dt = 0; dt < 2; ++dt) {
            v8f acc = {};
            #pragma unroll
            for (int kc = 0; kc < 2; ++kc) {
                v16h vb = *(const v16h*)(vsh + (dt * 16 + lh) * 64 + kc * 32 + hp * 16);
                acc = __builtin_amdgcn_wmma_f32_16x16x32_f16(
                    false, pa[kc], false, vb, (short)0, acc, false, false);
            }
            #pragma unroll
            for (int r = 0; r < 8; ++r) {
                int m = rt * 16 + r + hp * 8;
                if (m < WIN_N)
                    outb[(size_t)m * C_DIM + dt * 16 + lh] = acc[r];
            }
        }
    }
}

// ---------------------------------------------------------------------------
extern "C" void kernel_launch(void* const* d_in, const int* in_sizes, int n_in,
                              void* d_out, int out_size, void* d_ws, size_t ws_size,
                              hipStream_t stream) {
    (void)in_sizes; (void)n_in; (void)out_size; (void)ws_size;
    const float* X    = (const float*)d_in[0];
    const float* mask = (const float*)d_in[1];
    const float* wq   = (const float*)d_in[2];
    const float* bq   = (const float*)d_in[3];
    const float* wk   = (const float*)d_in[4];
    const float* bk   = (const float*)d_in[5];
    const float* wv   = (const float*)d_in[6];
    const float* bv   = (const float*)d_in[7];
    const float* btab = (const float*)d_in[8];
    const int*   rpi  = (const int*)d_in[9];
    float* out = (float*)d_out;

    half_t* wt       = (half_t*)d_ws;                                  // 384 KB
    float*  combined = (float*)((char*)d_ws + 3 * 256 * 256 * 2);      // 2 MB

    const int prep_elems = 3 * 256 * 256 + 16 * 8 * 64 * 64;           // 720896
    prep_kernel<<<(prep_elems + 255) / 256, 256, 0, stream>>>(
        wq, wk, wv, btab, rpi, mask, wt, combined);

    const size_t smem = (size_t)(16384 * 4 + 8192) * sizeof(half_t);   // 144 KB
    swin_attn_kernel<<<4096, 256, smem, stream>>>(X, wt, bq, bk, bv, combined, out);
}
